// ObjectDecoder_30081950941812
// MI455X (gfx1250) — matrix operand order repaired and checked by reading, not compiled, not run on hardware
//
#include <hip/hip_runtime.h>
#include <hip/hip_bf16.h>

// ---------------------------------------------------------------------------
// Types for CDNA5 WMMA
// ---------------------------------------------------------------------------
typedef __attribute__((ext_vector_type(16))) _Float16 v16h;
typedef __attribute__((ext_vector_type(8)))  float    v8f;
typedef __attribute__((ext_vector_type(4)))  int      v4i;

#define DMODEL 256
#define NHEADS 8
#define HD 32
#define NQ 900
#define BEV_H 200
#define BEV_W 200
#define NCLS 10
#define BS 4
#define NL 6
#define HWSZ (BEV_H * BEV_W)

__device__ __forceinline__ float sigmoidf_(float x) { return 1.f / (1.f + __expf(-x)); }
__device__ __forceinline__ float invsigf_(float x) {
    x = fminf(fmaxf(x, 0.f), 1.f);
    float a = fmaxf(x, 1e-5f);
    float b = fmaxf(1.f - x, 1e-5f);
    return logf(a / b);
}

// ---------------------------------------------------------------------------
// Generic batched GEMM via V_WMMA_F32_16X16X32_F16.
//   Out = act( alpha * A @ op(B) + bias + Res )
//   A: (M,K) f32, row stride lda
//   B: if bNK==1 -> stored (N,K) (weights, out = A @ B^T); read B[n*ldb + k]
//      if bNK==0 -> stored (K,N);                          read B[k*ldb + n]
//   batch z: zb=z/nH, zh=z%nH; per-operand offsets zb*s?b + zh*s?h
//
// Block tile 64(M) x 64(N) x 32(K); 256 threads = 8 waves.
// Wave (wm=wv>>1 in 0..3, wn=wv&1 in 0..1) computes a 16x32 strip:
// one A fragment reused across two B fragments -> 2 WMMAs per wave per K-step.
// LDS rows padded to 40 halves (80B): 16B-aligned, conflict-free ds_load_b128.
// ---------------------------------------------------------------------------
__global__ __launch_bounds__(256) void gemm_wmma_f16(
    const float* __restrict__ A, long sAb, long sAh, int lda,
    const float* __restrict__ Bm, long sBb, long sBh, int ldb, int bNK,
    const float* __restrict__ bias,
    const float* __restrict__ Res, long sRb, long sRh, int ldr,
    float* __restrict__ Out, long sOb, long sOh, int ldo,
    int M, int N, int K, int nH, float alpha, int act)
{
    int z  = blockIdx.z;
    int zb = z / nH, zh = z % nH;
    A  += zb * sAb + zh * sAh;
    Bm += zb * sBb + zh * sBh;
    Out += zb * sOb + zh * sOh;
    if (Res) Res += zb * sRb + zh * sRh;

    __shared__ _Float16 As[64][40];
    __shared__ _Float16 Bs[64][40];

    const int tid  = threadIdx.x;
    const int lane = tid & 31;
    const int wv   = tid >> 5;           // wave 0..7
    const int wm   = wv >> 1;            // wave row (0..3)
    const int wn   = wv & 1;             // wave col (0..1)
    const int grp  = lane >> 4;          // lane half-group
    const int r    = lane & 15;

    const int m0 = blockIdx.y * 64;
    const int n0 = blockIdx.x * 64;

    // staging assignment: thread -> (row 0..63, k-segment of 8)
    const int srow  = tid >> 2;
    const int skseg = (tid & 3) * 8;

    const bool interior = (m0 + 64 <= M) && (n0 + 64 <= N);

    v8f acc0 = {0.f, 0.f, 0.f, 0.f, 0.f, 0.f, 0.f, 0.f};
    v8f acc1 = {0.f, 0.f, 0.f, 0.f, 0.f, 0.f, 0.f, 0.f};

    for (int k0 = 0; k0 < K; k0 += 32) {
        union { _Float16 h[8]; v4i v; } ua, ub;
        const bool kfast = interior && (k0 + 32 <= K);
        if (kfast) {
            // fully in-bounds: unconditional contiguous loads (vectorizable)
            const float* ap = A + (long)(m0 + srow) * lda + (k0 + skseg);
            #pragma unroll
            for (int e = 0; e < 8; ++e) ua.h[e] = (_Float16)ap[e];
            if (bNK) {
                const float* bp = Bm + (long)(n0 + srow) * ldb + (k0 + skseg);
                #pragma unroll
                for (int e = 0; e < 8; ++e) ub.h[e] = (_Float16)bp[e];
            } else {
                #pragma unroll
                for (int e = 0; e < 8; ++e)
                    ub.h[e] = (_Float16)Bm[(long)(k0 + skseg + e) * ldb + (n0 + srow)];
            }
            if (k0 + 32 < K) {   // prefetch next K tile
                __builtin_prefetch(A + (long)(m0 + srow) * lda + (k0 + 32 + skseg), 0, 3);
                if (bNK)
                    __builtin_prefetch(Bm + (long)(n0 + srow) * ldb + (k0 + 32 + skseg), 0, 3);
            }
        } else {
            // boundary: clamped addresses (unconditional loads) + select-to-zero
            const int gm = m0 + srow, gn = n0 + srow;
            const int gmc = min(gm, M - 1), gnc = min(gn, N - 1);
            #pragma unroll
            for (int e = 0; e < 8; ++e) {
                int gk  = k0 + skseg + e;
                int gkc = min(gk, K - 1);
                float av = A[(long)gmc * lda + gkc];
                ua.h[e] = (_Float16)((gm < M && gk < K) ? av : 0.f);
                float bv = bNK ? Bm[(long)gnc * ldb + gkc]
                               : Bm[(long)gkc * ldb + gnc];
                ub.h[e] = (_Float16)((gn < N && gk < K) ? bv : 0.f);
            }
        }
        *(v4i*)&As[srow][skseg] = ua.v;
        *(v4i*)&Bs[srow][skseg] = ub.v;
        __syncthreads();

        // Fragment loads per ISA 16-bit layouts (7.12.2), b128 pairs from LDS
        const int ar  = wm * 16 + r;        // A row in tile
        const int bc0 = wn * 32 + r;        // B col (first 16x16)
        const int bc1 = bc0 + 16;           // B col (second 16x16)
        union { v4i p[2]; v16h v; } af, bf0, bf1;
        af.p[0]  = *(const v4i*)&As[ar][grp * 8];        // K = grp*8 + 0..7
        af.p[1]  = *(const v4i*)&As[ar][16 + grp * 8];   // K = 16 + grp*8 + 0..7
        bf0.p[0] = *(const v4i*)&Bs[bc0][grp * 8];       // K = grp*8 + 0..7
        bf0.p[1] = *(const v4i*)&Bs[bc0][16 + grp * 8];  // K = 16 + grp*8 + 0..7
        bf1.p[0] = *(const v4i*)&Bs[bc1][grp * 8];
        bf1.p[1] = *(const v4i*)&Bs[bc1][16 + grp * 8];

        acc0 = __builtin_amdgcn_wmma_f32_16x16x32_f16(
            false, af.v, false, bf0.v, (short)0, acc0, false, false);
        acc1 = __builtin_amdgcn_wmma_f32_16x16x32_f16(
            false, af.v, false, bf1.v, (short)0, acc1, false, false);
        __syncthreads();
    }

    const int rowb = m0 + wm * 16 + grp * 8;
    const int c0   = n0 + wn * 32 + r;
    const int c1   = c0 + 16;
    const float bb0 = (bias && c0 < N) ? bias[c0] : 0.f;
    const float bb1 = (bias && c1 < N) ? bias[c1] : 0.f;
    #pragma unroll
    for (int e = 0; e < 8; ++e) {
        int row = rowb + e;
        if (row < M) {
            if (c0 < N) {
                float v = acc0[e] * alpha + bb0;
                if (Res) v += Res[(long)row * ldr + c0];
                if (act == 1)      v = fmaxf(v, 0.f);
                else if (act == 2) v = sigmoidf_(v);
                Out[(long)row * ldo + c0] = v;
            }
            if (c1 < N) {
                float v = acc1[e] * alpha + bb1;
                if (Res) v += Res[(long)row * ldr + c1];
                if (act == 1)      v = fmaxf(v, 0.f);
                else if (act == 2) v = sigmoidf_(v);
                Out[(long)row * ldo + c1] = v;
            }
        }
    }
}

// ---------------------------------------------------------------------------
// LayerNorm over last dim (=256). One block (256 threads) per row.
// out = LN(x [+ res]) * w + b, optional relu.
// ---------------------------------------------------------------------------
__global__ __launch_bounds__(256) void ln256_kernel(
    const float* __restrict__ x, const float* __restrict__ res,
    const float* __restrict__ w, const float* __restrict__ b,
    float* __restrict__ out, int act)
{
    __shared__ float red[256];
    const int row = blockIdx.x, tid = threadIdx.x;
    const long base = (long)row * DMODEL + tid;
    float v = x[base] + (res ? res[base] : 0.f);
    red[tid] = v; __syncthreads();
    for (int o = 128; o > 0; o >>= 1) { if (tid < o) red[tid] += red[tid + o]; __syncthreads(); }
    float mean = red[0] * (1.f / DMODEL);
    __syncthreads();
    float dv = v - mean;
    red[tid] = dv * dv; __syncthreads();
    for (int o = 128; o > 0; o >>= 1) { if (tid < o) red[tid] += red[tid + o]; __syncthreads(); }
    float var = red[0] * (1.f / DMODEL);
    float y = dv * rsqrtf(var + 1e-5f) * w[tid] + b[tid];
    if (act == 1) y = fmaxf(y, 0.f);
    out[base] = y;
}

// ---------------------------------------------------------------------------
// Row softmax, one block per row, arbitrary width (in==out allowed)
// ---------------------------------------------------------------------------
__global__ void softmax_rows_kernel(const float* __restrict__ in,
                                    float* __restrict__ out, int W)
{
    __shared__ float red[256];
    const int row = blockIdx.x, tid = threadIdx.x, nt = blockDim.x;
    const long base = (long)row * W;
    float m = -3.4e38f;
    for (int j = tid; j < W; j += nt) m = fmaxf(m, in[base + j]);
    red[tid] = m; __syncthreads();
    for (int o = nt >> 1; o > 0; o >>= 1) { if (tid < o) red[tid] = fmaxf(red[tid], red[tid + o]); __syncthreads(); }
    m = red[0]; __syncthreads();
    float s = 0.f;
    for (int j = tid; j < W; j += nt) { float e = __expf(in[base + j] - m); out[base + j] = e; s += e; }
    red[tid] = s; __syncthreads();
    for (int o = nt >> 1; o > 0; o >>= 1) { if (tid < o) red[tid] += red[tid + o]; __syncthreads(); }
    float inv = 1.f / red[0];
    for (int j = tid; j < W; j += nt) out[base + j] *= inv;
}

// out[i] = a[i] + bcast[i % per]   (broadcast query_pos over batch)
__global__ void add_bcast_kernel(const float* __restrict__ a,
                                 const float* __restrict__ bc,
                                 float* __restrict__ out, long total, long per)
{
    long i = (long)blockIdx.x * blockDim.x + threadIdx.x;
    if (i < total) out[i] = a[i] + bc[i % per];
}

// qpos (900,256) and q0 (4,900,256) from query_embed (900,512)
__global__ void init_qposq0_kernel(const float* __restrict__ qe,
                                   float* __restrict__ qpos,
                                   float* __restrict__ q0, long bqd, long qd)
{
    long i = (long)blockIdx.x * blockDim.x + threadIdx.x;
    if (i < qd) {
        long n = i / DMODEL, d = i % DMODEL;
        qpos[i] = qe[n * 2 * DMODEL + d];
    }
    if (i < bqd) {
        long rest = i % qd;
        long n = rest / DMODEL, d = rest % DMODEL;
        q0[i] = qe[n * 2 * DMODEL + DMODEL + d];
    }
}

// ---------------------------------------------------------------------------
// Deformable bilinear sampling for one batch. grid (NQ, NHEADS), block 32 (=HD)
//   vgrid : (HW, 256) projected values for this batch
//   off   : (NQ, 8*4*2)   aw : (NQ, 8*4)   ref : (NQ, 3)
//   out   : (NQ, 256)
// ---------------------------------------------------------------------------
__global__ __launch_bounds__(32) void deform_sample_kernel(
    const float* __restrict__ vgrid, const float* __restrict__ off,
    const float* __restrict__ aw, const float* __restrict__ ref,
    float* __restrict__ out)
{
    const int n = blockIdx.x, h = blockIdx.y, d = threadIdx.x;
    const float rx = ref[n * 3 + 0];
    const float ry = ref[n * 3 + 1];
    float acc = 0.f;
    #pragma unroll
    for (int p = 0; p < 4; ++p) {
        float ox = off[n * 64 + h * 8 + p * 2 + 0];
        float oy = off[n * 64 + h * 8 + p * 2 + 1];
        float wp = aw[n * 32 + h * 4 + p];
        float lx = rx + ox * (1.f / BEV_W);
        float ly = ry + oy * (1.f / BEV_H);
        float X = lx * BEV_W - 0.5f;
        float Y = ly * BEV_H - 0.5f;
        float x0f = floorf(X), y0f = floorf(Y);
        int x0 = (int)x0f, y0 = (int)y0f;
        float fx = X - x0f, fy = Y - y0f;
        #pragma unroll
        for (int dy = 0; dy < 2; ++dy) {
            #pragma unroll
            for (int dx = 0; dx < 2; ++dx) {
                int xi = x0 + dx, yi = y0 + dy;
                bool valid = (xi >= 0) && (xi < BEV_W) && (yi >= 0) && (yi < BEV_H);
                int xc = min(max(xi, 0), BEV_W - 1);
                int yc = min(max(yi, 0), BEV_H - 1);
                long lin = (long)yc * BEV_W + xc;
                float g = vgrid[lin * DMODEL + h * HD + d];
                float wx = dx ? fx : (1.f - fx);
                float wy = dy ? fy : (1.f - fy);
                acc += wp * (valid ? wx * wy : 0.f) * g;
            }
        }
    }
    out[(long)n * DMODEL + h * HD + d] = acc;
}

// new_ref = [sig(t0+inv(r0)), sig(t1+inv(r1)), sig(t4+inv(r2))]
__global__ void ref_update_kernel(const float* __restrict__ tmp,
                                  const float* __restrict__ ref,
                                  float* __restrict__ nref, int total)
{
    int i = blockIdx.x * blockDim.x + threadIdx.x;
    if (i >= total) return;
    const float* t = tmp + (long)i * 10;
    const float* r = ref + (long)i * 3;
    float* o = nref + (long)i * 3;
    o[0] = sigmoidf_(t[0] + invsigf_(r[0]));
    o[1] = sigmoidf_(t[1] + invsigf_(r[1]));
    o[2] = sigmoidf_(t[4] + invsigf_(r[2]));
}

// coords into out[..., 10:20] for one layer; out base = d_out + layer*BS*NQ*20
__global__ void coord_kernel(const float* __restrict__ tmp,
                             const float* __restrict__ refprev,
                             float* __restrict__ out, int total)
{
    int i = blockIdx.x * blockDim.x + threadIdx.x;
    if (i >= total) return;
    const float* t = tmp + (long)i * 10;
    const float* r = refprev + (long)i * 3;
    float* o = out + (long)i * 20 + 10;
    float r0 = invsigf_(r[0]), r1 = invsigf_(r[1]), r2 = invsigf_(r[2]);
    o[0] = sigmoidf_(t[0] + r0) * 102.4f - 51.2f;
    o[1] = sigmoidf_(t[1] + r1) * 102.4f - 51.2f;
    o[2] = t[2];
    o[3] = t[3];
    o[4] = sigmoidf_(t[4] + r2) * 8.f - 5.f;
    o[5] = t[5]; o[6] = t[6]; o[7] = t[7]; o[8] = t[8]; o[9] = t[9];
}

// ---------------------------------------------------------------------------
// Host orchestration
// ---------------------------------------------------------------------------
static inline void gemm(hipStream_t s,
    const float* A, long sAb, long sAh, int lda,
    const float* B, long sBb, long sBh, int ldb, int bNK,
    const float* bias,
    const float* Res, long sRb, long sRh, int ldr,
    float* Out, long sOb, long sOh, int ldo,
    int M, int N, int K, int batches, int nH, float alpha, int act)
{
    dim3 g((N + 63) / 64, (M + 63) / 64, batches);
    gemm_wmma_f16<<<g, 256, 0, s>>>(A, sAb, sAh, lda, B, sBb, sBh, ldb, bNK,
                                    bias, Res, sRb, sRh, ldr,
                                    Out, sOb, sOh, ldo, M, N, K, nH, alpha, act);
}

extern "C" void kernel_launch(void* const* d_in, const int* in_sizes, int n_in,
                              void* d_out, int out_size, void* d_ws, size_t ws_size,
                              hipStream_t stream)
{
    (void)in_sizes; (void)n_in; (void)out_size; (void)ws_size;
    const float* bev      = (const float*)d_in[0];
    const float* qe       = (const float*)d_in[1];
    const float* ref_w    = (const float*)d_in[2];
    const float* ref_b    = (const float*)d_in[3];
    const float* sa_in_w  = (const float*)d_in[4];
    const float* sa_in_b  = (const float*)d_in[5];
    const float* sa_out_w = (const float*)d_in[6];
    const float* sa_out_b = (const float*)d_in[7];
    const float* ln_w     = (const float*)d_in[8];
    const float* ln_b     = (const float*)d_in[9];
    const float* off_w    = (const float*)d_in[10];
    const float* off_b    = (const float*)d_in[11];
    const float* aw_w     = (const float*)d_in[12];
    const float* aw_b     = (const float*)d_in[13];
    const float* vproj_w  = (const float*)d_in[14];
    const float* vproj_b  = (const float*)d_in[15];
    const float* oproj_w  = (const float*)d_in[16];
    const float* oproj_b  = (const float*)d_in[17];
    const float* ffn_w1   = (const float*)d_in[18];
    const float* ffn_b1   = (const float*)d_in[19];
    const float* ffn_w2   = (const float*)d_in[20];
    const float* ffn_b2   = (const float*)d_in[21];
    const float* reg_w1   = (const float*)d_in[22];
    const float* reg_b1   = (const float*)d_in[23];
    const float* reg_w2   = (const float*)d_in[24];
    const float* reg_b2   = (const float*)d_in[25];
    const float* reg_w3   = (const float*)d_in[26];
    const float* reg_b3   = (const float*)d_in[27];
    const float* cls_w1   = (const float*)d_in[28];
    const float* cls_b1   = (const float*)d_in[29];
    const float* cls_w2   = (const float*)d_in[30];
    const float* cls_b2   = (const float*)d_in[31];
    const float* cls_w3   = (const float*)d_in[32];
    const float* cls_b3   = (const float*)d_in[33];
    const float* cls_ln_w = (const float*)d_in[34];
    const float* cls_ln_b = (const float*)d_in[35];
    float* outp = (float*)d_out;

    const long QD  = (long)NQ * DMODEL;      // 230400
    const long BQD = (long)BS * QD;          // 921600
    const int  ROWS = BS * NQ;               // 3600

    // bump allocator over workspace
    float* w = (float*)d_ws;
    size_t woff = 0;
    auto alloc = [&](size_t n) { float* p = w + woff; woff += n; return p; };

    float* hs      = alloc(6 * BQD);
    float* regtmps = alloc(6L * ROWS * 10);
    float* initref = alloc((size_t)ROWS * 3);
    float* refs    = alloc(6L * ROWS * 3);
    float* qpos    = alloc(QD);
    float* q0      = alloc(BQD);
    float* x1      = alloc(BQD);
    float* qk      = alloc((long)BS * NQ * 512);
    float* vat     = alloc(BQD);
    float* scores  = alloc(8L * NQ * NQ);
    float* ctx     = alloc(BQD);
    float* sa      = alloc(BQD);
    float* qA      = alloc(BQD);
    float* x2      = alloc(BQD);
    float* offb    = alloc((size_t)ROWS * 64);
    float* awb     = alloc((size_t)ROWS * 32);
    float* vgrid   = alloc((long)HWSZ * DMODEL);
    float* samp    = alloc(BQD);
    float* ca      = alloc(BQD);
    float* qB      = alloc(BQD);
    float* ffh     = alloc((long)BS * NQ * 512);
    float* ffo     = alloc(BQD);
    float* t1      = alloc(BQD);
    float* t2      = alloc(BQD);

    // init qpos / q0
    init_qposq0_kernel<<<(unsigned)((BQD + 255) / 256), 256, 0, stream>>>(qe, qpos, q0, BQD, QD);
    // init_ref = sigmoid(qpos @ ref_w^T + ref_b), broadcast to batch via zero A-stride
    gemm(stream, qpos, 0, 0, DMODEL, ref_w, 0, 0, DMODEL, 1, ref_b,
         nullptr, 0, 0, 0, initref, (long)NQ * 3, 0, 3,
         NQ, 3, DMODEL, BS, 1, 1.f, 2);

    const float* qcur = q0;
    const float* refcur = initref;
    const float scale = 0.17677669529663687f; // 1/sqrt(32)

    for (int i = 0; i < NL; ++i) {
        const float* inw = sa_in_w + (long)i * 3 * DMODEL * DMODEL;
        const float* inb = sa_in_b + (long)i * 3 * DMODEL;
        // x1 = q + qpos
        add_bcast_kernel<<<(unsigned)((BQD + 255) / 256), 256, 0, stream>>>(qcur, qpos, x1, BQD, QD);
        // qk = x1 @ in_w[0:512]^T + b  -> (B,900,512)  cols [0:256)=Q [256:512)=K
        gemm(stream, x1, QD, 0, DMODEL, inw, 0, 0, DMODEL, 1, inb,
             nullptr, 0, 0, 0, qk, (long)NQ * 512, 0, 512,
             NQ, 2 * DMODEL, DMODEL, BS, 1, 1.f, 0);
        // vat = q @ in_w[512:768]^T + b
        gemm(stream, qcur, QD, 0, DMODEL, inw + 2L * DMODEL * DMODEL, 0, 0, DMODEL, 1,
             inb + 2 * DMODEL, nullptr, 0, 0, 0, vat, QD, 0, DMODEL,
             NQ, DMODEL, DMODEL, BS, 1, 1.f, 0);
        // attention per batch (reuse one 8x900x900 score buffer)
        for (int b = 0; b < BS; ++b) {
            const float* qkb = qk + (long)b * NQ * 512;
            gemm(stream, qkb, 0, HD, 512, qkb + DMODEL, 0, HD, 512, 1, nullptr,
                 nullptr, 0, 0, 0, scores, 0, (long)NQ * NQ, NQ,
                 NQ, NQ, HD, NHEADS, NHEADS, scale, 0);
            softmax_rows_kernel<<<NHEADS * NQ, 256, 0, stream>>>(scores, scores, NQ);
            gemm(stream, scores, 0, (long)NQ * NQ, NQ,
                 vat + (long)b * QD, 0, HD, DMODEL, 0, nullptr,
                 nullptr, 0, 0, 0, ctx + (long)b * QD, 0, HD, DMODEL,
                 NQ, HD, NQ, NHEADS, NHEADS, 1.f, 0);
        }
        // sa = ctx @ out_w^T + out_b
        gemm(stream, ctx, QD, 0, DMODEL, sa_out_w + (long)i * DMODEL * DMODEL, 0, 0, DMODEL, 1,
             sa_out_b + (long)i * DMODEL, nullptr, 0, 0, 0, sa, QD, 0, DMODEL,
             NQ, DMODEL, DMODEL, BS, 1, 1.f, 0);
        // qA = LN(q + sa)
        ln256_kernel<<<ROWS, 256, 0, stream>>>(qcur, sa,
            ln_w + ((long)i * 3 + 0) * DMODEL, ln_b + ((long)i * 3 + 0) * DMODEL, qA, 0);
        // x2 = qA + qpos
        add_bcast_kernel<<<(unsigned)((BQD + 255) / 256), 256, 0, stream>>>(qA, qpos, x2, BQD, QD);
        // offsets / attention weights
        gemm(stream, x2, QD, 0, DMODEL, off_w + (long)i * 64 * DMODEL, 0, 0, DMODEL, 1,
             off_b + (long)i * 64, nullptr, 0, 0, 0, offb, (long)NQ * 64, 0, 64,
             NQ, 64, DMODEL, BS, 1, 1.f, 0);
        gemm(stream, x2, QD, 0, DMODEL, aw_w + (long)i * 32 * DMODEL, 0, 0, DMODEL, 1,
             aw_b + (long)i * 32, nullptr, 0, 0, 0, awb, (long)NQ * 32, 0, 32,
             NQ, 32, DMODEL, BS, 1, 1.f, 0);
        softmax_rows_kernel<<<ROWS * NHEADS, 32, 0, stream>>>(awb, awb, 4);
        // per batch: value projection over BEV grid then bilinear sampling
        for (int b = 0; b < BS; ++b) {
            gemm(stream, bev + (long)b * HWSZ * DMODEL, 0, 0, DMODEL,
                 vproj_w + (long)i * DMODEL * DMODEL, 0, 0, DMODEL, 1,
                 vproj_b + (long)i * DMODEL, nullptr, 0, 0, 0,
                 vgrid, 0, 0, DMODEL, HWSZ, DMODEL, DMODEL, 1, 1, 1.f, 0);
            deform_sample_kernel<<<dim3(NQ, NHEADS), 32, 0, stream>>>(
                vgrid, offb + (long)b * NQ * 64, awb + (long)b * NQ * 32,
                refcur + (long)b * NQ * 3, samp + (long)b * QD);
        }
        // ca = samp @ oproj^T + oproj_b + identity(qA)
        gemm(stream, samp, QD, 0, DMODEL, oproj_w + (long)i * DMODEL * DMODEL, 0, 0, DMODEL, 1,
             oproj_b + (long)i * DMODEL, qA, QD, 0, DMODEL, ca, QD, 0, DMODEL,
             NQ, DMODEL, DMODEL, BS, 1, 1.f, 0);
        // qB = LN(ca)
        ln256_kernel<<<ROWS, 256, 0, stream>>>(ca, nullptr,
            ln_w + ((long)i * 3 + 1) * DMODEL, ln_b + ((long)i * 3 + 1) * DMODEL, qB, 0);
        // FFN
        gemm(stream, qB, QD, 0, DMODEL, ffn_w1 + (long)i * 512 * DMODEL, 0, 0, DMODEL, 1,
             ffn_b1 + (long)i * 512, nullptr, 0, 0, 0, ffh, (long)NQ * 512, 0, 512,
             NQ, 512, DMODEL, BS, 1, 1.f, 1);
        gemm(stream, ffh, (long)NQ * 512, 0, 512, ffn_w2 + (long)i * DMODEL * 512, 0, 0, 512, 1,
             ffn_b2 + (long)i * DMODEL, nullptr, 0, 0, 0, ffo, QD, 0, DMODEL,
             NQ, DMODEL, 512, BS, 1, 1.f, 0);
        float* hsi = hs + (long)i * BQD;
        ln256_kernel<<<ROWS, 256, 0, stream>>>(qB, ffo,
            ln_w + ((long)i * 3 + 2) * DMODEL, ln_b + ((long)i * 3 + 2) * DMODEL, hsi, 0);
        // reg branch -> tmp10 (cached per layer, reused by output heads)
        gemm(stream, hsi, QD, 0, DMODEL, reg_w1 + (long)i * DMODEL * DMODEL, 0, 0, DMODEL, 1,
             reg_b1 + (long)i * DMODEL, nullptr, 0, 0, 0, t1, QD, 0, DMODEL,
             NQ, DMODEL, DMODEL, BS, 1, 1.f, 1);
        gemm(stream, t1, QD, 0, DMODEL, reg_w2 + (long)i * DMODEL * DMODEL, 0, 0, DMODEL, 1,
             reg_b2 + (long)i * DMODEL, nullptr, 0, 0, 0, t2, QD, 0, DMODEL,
             NQ, DMODEL, DMODEL, BS, 1, 1.f, 1);
        float* tmpi = regtmps + (long)i * ROWS * 10;
        gemm(stream, t2, QD, 0, DMODEL, reg_w3 + (long)i * 10 * DMODEL, 0, 0, DMODEL, 1,
             reg_b3 + (long)i * 10, nullptr, 0, 0, 0, tmpi, (long)NQ * 10, 0, 10,
             NQ, 10, DMODEL, BS, 1, 1.f, 0);
        // ref refinement
        float* refi = refs + (long)i * ROWS * 3;
        ref_update_kernel<<<(ROWS + 255) / 256, 256, 0, stream>>>(tmpi, refcur, refi, ROWS);
        refcur = refi;
        qcur = hsi;
    }

    // output heads
    for (int i = 0; i < NL; ++i) {
        const float* xi = hs + (long)i * BQD;
        // cls branch: relu(LN(x@W1+b1)) -> relu(LN(.@W2+b2)) -> @W3+b3
        gemm(stream, xi, QD, 0, DMODEL, cls_w1 + (long)i * DMODEL * DMODEL, 0, 0, DMODEL, 1,
             cls_b1 + (long)i * DMODEL, nullptr, 0, 0, 0, t1, QD, 0, DMODEL,
             NQ, DMODEL, DMODEL, BS, 1, 1.f, 0);
        ln256_kernel<<<ROWS, 256, 0, stream>>>(t1, nullptr,
            cls_ln_w + ((long)i * 2 + 0) * DMODEL, cls_ln_b + ((long)i * 2 + 0) * DMODEL, t2, 1);
        gemm(stream, t2, QD, 0, DMODEL, cls_w2 + (long)i * DMODEL * DMODEL, 0, 0, DMODEL, 1,
             cls_b2 + (long)i * DMODEL, nullptr, 0, 0, 0, t1, QD, 0, DMODEL,
             NQ, DMODEL, DMODEL, BS, 1, 1.f, 0);
        ln256_kernel<<<ROWS, 256, 0, stream>>>(t1, nullptr,
            cls_ln_w + ((long)i * 2 + 1) * DMODEL, cls_ln_b + ((long)i * 2 + 1) * DMODEL, t2, 1);
        float* outl = outp + (long)i * BS * NQ * 20;
        gemm(stream, t2, QD, 0, DMODEL, cls_w3 + (long)i * NCLS * DMODEL, 0, 0, DMODEL, 1,
             cls_b3 + (long)i * NCLS, nullptr, 0, 0, 0, outl, (long)NQ * 20, 0, 20,
             NQ, NCLS, DMODEL, BS, 1, 1.f, 0);
        // coords from cached reg tmp + previous-layer ref
        const float* rsel = (i == 0) ? initref : (refs + (long)(i - 1) * ROWS * 3);
        coord_kernel<<<(ROWS + 255) / 256, 256, 0, stream>>>(
            regtmps + (long)i * ROWS * 10, rsel, outl, ROWS);
    }
}
